// DbrxBlock_53352083751571
// MI455X (gfx1250) — matrix operand order repaired and checked by reading, not compiled, not run on hardware
//
#include <hip/hip_runtime.h>
#include <stdint.h>

// ---------------------------------------------------------------------------
// DBRX MoE FFN forward for MI455X (gfx1250, wave32).
// Top-2 grouped-GEMM formulation; f16 WMMA (v_wmma_f32_16x16x32_f16) with
// fp32 accumulate; fp32 weights converted to f16 tile-wise on the fly.
// A tiles gathered into LDS via global_load_async_to_lds_b128 (ASYNCcnt),
// double-buffered so async copy of step i+1 overlaps WMMA of step i.
// Workspace requirement: ~88.2 MB.
// ---------------------------------------------------------------------------

typedef __attribute__((ext_vector_type(16))) _Float16 v16h;
typedef __attribute__((ext_vector_type(8)))  _Float16 v8h;
typedef __attribute__((ext_vector_type(4)))  _Float16 v4h;
typedef __attribute__((ext_vector_type(8)))  float    v8f;

#define TT 2048      // tokens (B*S)
#define HH 2048      // hidden
#define FF 5632      // ffn dim
#define EE 8         // experts
#define NSLOT (TT*2) // total (token, expert) slots == T*TOP_K

// ---- workspace layout (bytes) ----
#define OFF_XH     ((size_t)0)                         // [TT][HH] f16      8388608
#define OFF_TOPI   ((size_t)8388608)                   // [TT][2]  int     16384
#define OFF_TOPW   (OFF_TOPI + 16384)                  // [TT][2]  f32     16384
#define OFF_GLIST  (OFF_TOPW + 16384)                  // [NSLOT]  int     16384
#define OFF_WSLOT  (OFF_GLIST + 16384)                 // [NSLOT]  f32     16384
#define OFF_SLOTOF (OFF_WSLOT + 16384)                 // [TT][2]  int     16384
#define OFF_SEG    (OFF_SLOTOF + 16384)                // base[8], cnt[8]  64
#define OFF_ACTS   (OFF_SEG + 256)                     // [NSLOT][FF] f16  46137344
#define OFF_YBUF   (OFF_ACTS + (size_t)NSLOT*FF*2)     // [NSLOT][HH] f32  33554432
// total = 88,162,560 bytes

__device__ __forceinline__ v16h cat16(v8h lo, v8h hi) {
  return __builtin_shufflevector(lo, hi, 0,1,2,3,4,5,6,7,8,9,10,11,12,13,14,15);
}

// A fragment, 16-bit A 16x32 (ISA 7.12.2): lanes 0-15 hold M=lane, K={0..7,16..23};
// lanes 16-31 hold M=lane-16, K={8..15,24..31}.
__device__ __forceinline__ v16h ldfragA(const _Float16 (&sA)[64][32], int m, int k8) {
  v8h lo = *(const v8h*)&sA[m][k8];
  v8h hi = *(const v8h*)&sA[m][k8 + 16];
  return cat16(lo, hi);
}
// B fragment, 16-bit B 32x16: lane holds N=lane&15, K={0..7,16..23} (lanes 0-15)
// or K={8..15,24..31} (lanes 16-31).
__device__ __forceinline__ v16h ldfragB(const _Float16 (&sB)[64][32], int n, int kb) {
  v8h lo = *(const v8h*)&sB[n][kb];
  v8h hi = *(const v8h*)&sB[n][kb + 16];
  return cat16(lo, hi);
}

__device__ __forceinline__ v8f wmma16(v16h a, v16h b, v8f c) {
  return __builtin_amdgcn_wmma_f32_16x16x32_f16(false, a, false, b, (short)0, c,
                                                false, false);
}

__device__ __forceinline__ void async_copy16(unsigned lds_off, uint64_t gaddr) {
  asm volatile("global_load_async_to_lds_b128 %0, %1, off"
               :: "v"(lds_off), "v"(gaddr) : "memory");
}
__device__ __forceinline__ void wait_async0() {
  asm volatile("s_wait_asynccnt 0" ::: "memory");
}

__device__ __forceinline__ v8h cvt8(const float4& u, const float4& v) {
  return (v8h){(_Float16)u.x, (_Float16)u.y, (_Float16)u.z, (_Float16)u.w,
               (_Float16)v.x, (_Float16)v.y, (_Float16)v.z, (_Float16)v.w};
}

// silu(p)*q with raw v_rcp_f32 (1 ulp) instead of IEEE divide.
__device__ __forceinline__ float silu_gate(float p, float q) {
  return p * q * __builtin_amdgcn_rcpf(1.f + __expf(-p));
}

// ---------------------------------------------------------------------------
// K0: x fp32 -> f16 (A operand of GEMM1, reused F=5632 times -> convert once)
// ---------------------------------------------------------------------------
__global__ void k_f2h(const float* __restrict__ x, _Float16* __restrict__ xh) {
  int i = (blockIdx.x * 256 + threadIdx.x) * 4;
  float4 v = *(const float4*)(x + i);
  v4h o = {(_Float16)v.x, (_Float16)v.y, (_Float16)v.z, (_Float16)v.w};
  *(v4h*)(xh + i) = o;
}

// ---------------------------------------------------------------------------
// K1: router. One wave per token: 8 dots of length H, softmax, top-2, L1 renorm.
// Writes softmax weights (second reference output) and per-token top-2.
// ---------------------------------------------------------------------------
__global__ __launch_bounds__(256) void k_router(
    const float* __restrict__ x, const float* __restrict__ Wr,
    float* __restrict__ wout, int* __restrict__ topi, float* __restrict__ topw) {
  int wave = threadIdx.x >> 5, lane = threadIdx.x & 31;
  int t = blockIdx.x * 8 + wave;
  const float* xr = x + (size_t)t * HH;
  float acc[EE];
#pragma unroll
  for (int e = 0; e < EE; e++) acc[e] = 0.f;
  for (int h = lane; h < HH; h += 32) {
    float xv = xr[h];
#pragma unroll
    for (int e = 0; e < EE; e++) acc[e] += xv * Wr[e * HH + h];
  }
#pragma unroll
  for (int e = 0; e < EE; e++)
#pragma unroll
    for (int off = 16; off > 0; off >>= 1)
      acc[e] += __shfl_xor(acc[e], off, 32);
  if (lane == 0) {
    float m = acc[0];
#pragma unroll
    for (int e = 1; e < EE; e++) m = fmaxf(m, acc[e]);
    float w[EE]; float s = 0.f;
#pragma unroll
    for (int e = 0; e < EE; e++) { w[e] = __expf(acc[e] - m); s += w[e]; }
    float inv = 1.f / s;
#pragma unroll
    for (int e = 0; e < EE; e++) { w[e] *= inv; wout[t * EE + e] = w[e]; }
    int e0 = 0; float b0 = w[0];
#pragma unroll
    for (int e = 1; e < EE; e++) if (w[e] > b0) { b0 = w[e]; e0 = e; }
    int e1 = -1; float b1 = -1.f;
#pragma unroll
    for (int e = 0; e < EE; e++) if (e != e0 && w[e] > b1) { b1 = w[e]; e1 = e; }
    float rs = 1.f / (b0 + b1);
    topi[t * 2] = e0;  topi[t * 2 + 1] = e1;
    topw[t * 2] = b0 * rs; topw[t * 2 + 1] = b1 * rs;
  }
}

// ---------------------------------------------------------------------------
// K2: deterministic expert compaction. Wave e builds expert e's contiguous
// slot segment via ballot/popc prefix (token order preserved).
// ---------------------------------------------------------------------------
__global__ __launch_bounds__(256) void k_build(
    const int* __restrict__ topi, const float* __restrict__ topw,
    int* __restrict__ glist, float* __restrict__ wslot,
    int* __restrict__ slotof, int* __restrict__ segmeta) {
  __shared__ int scnt[EE];
  __shared__ int sbase[EE];
  int e = threadIdx.x >> 5;
  int lane = threadIdx.x & 31;
  unsigned lmask = (1u << lane) - 1u;
  int c = 0;
  for (int t0 = 0; t0 < TT; t0 += 32) {
    int t = t0 + lane;
    int i0 = topi[2 * t], i1 = topi[2 * t + 1];
    unsigned m = (unsigned)__ballot((i0 == e) || (i1 == e));
    c += __popc(m);
  }
  if (lane == 0) scnt[e] = c;
  __syncthreads();
  if (threadIdx.x == 0) {
    int a = 0;
    for (int i = 0; i < EE; i++) { sbase[i] = a; a += scnt[i]; }
  }
  __syncthreads();
  int pos = sbase[e];
  for (int t0 = 0; t0 < TT; t0 += 32) {
    int t = t0 + lane;
    int i0 = topi[2 * t], i1 = topi[2 * t + 1];
    int k = (i0 == e) ? 0 : ((i1 == e) ? 1 : -1);
    unsigned m = (unsigned)__ballot(k >= 0);
    if (k >= 0) {
      int p = pos + __popc(m & lmask);
      glist[p] = t;
      wslot[p] = topw[2 * t + k];
      slotof[2 * t + k] = p;
    }
    pos += __popc(m);
  }
  if (lane == 0) { segmeta[e] = sbase[e]; segmeta[EE + e] = scnt[e]; }
}

// ---------------------------------------------------------------------------
// K3: GEMM1  acts = silu(x@w1[e]^T) * (x@v1[e]^T), grouped per expert.
// Block: 64 slots x 64 F-cols, 8 waves, K-step 32 over H, double-buffered LDS.
// ---------------------------------------------------------------------------
__global__ __launch_bounds__(256) void k_gemm1(
    const _Float16* __restrict__ xh, const float* __restrict__ w1,
    const float* __restrict__ v1, const int* __restrict__ glist,
    const int* __restrict__ segmeta, _Float16* __restrict__ acts) {
  int e = blockIdx.y;
  int cnt = segmeta[EE + e];
  int r0 = blockIdx.z * 64;
  if (r0 >= cnt) return;
  int base = segmeta[e];
  int f0 = blockIdx.x * 64;

  __shared__ _Float16 sA[2][64][32];
  __shared__ _Float16 sB1[2][64][32];
  __shared__ _Float16 sB2[2][64][32];

  int tid = threadIdx.x;
  int wv = tid >> 5, lane = tid & 31;
  int mbase = (wv & 3) * 16;
  int nbase = (wv >> 2) * 32;

  // A staging: thread -> (row, 16B segment); rows gathered through glist.
  int arow = tid >> 2;
  int aseg = (tid & 3) * 8;                 // in halfs
  int grow = r0 + arow;
  int tok = glist[base + (grow < cnt ? grow : cnt - 1)];
  const _Float16* asrc = xh + (size_t)tok * HH + aseg;
  unsigned aldst0 = (unsigned)(size_t)&sA[0][arow][aseg];
  unsigned aldst1 = (unsigned)(size_t)&sA[1][arow][aseg];

  // B staging: thread -> (row, 8 floats), convert fp32 -> f16.
  int brow = tid >> 2;
  int bc = (tid & 3) * 8;
  const float* b1src = w1 + ((size_t)e * FF + (f0 + brow)) * HH + bc;
  const float* b2src = v1 + ((size_t)e * FF + (f0 + brow)) * HH + bc;

  auto stage = [&](int k0, int buf) {
    async_copy16(buf ? aldst1 : aldst0, (uint64_t)(uintptr_t)(asrc + k0));
    const float4* p = (const float4*)(b1src + k0);
    *(v8h*)&sB1[buf][brow][bc] = cvt8(p[0], p[1]);
    p = (const float4*)(b2src + k0);
    *(v8h*)&sB2[buf][brow][bc] = cvt8(p[0], p[1]);
  };

  v8f aP0 = {}, aP1 = {}, aQ0 = {}, aQ1 = {};

  int fm = mbase + (lane & 15);
  int fk = (lane >> 4) * 8;
  int fn = nbase + (lane & 15);
  int fkb = (lane >> 4) * 8;

  const int NK = HH / 32;
  stage(0, 0);
  for (int i = 0; i < NK; ++i) {
    wait_async0();           // own async writes for buf (i&1) complete
    __syncthreads();         // all waves' writes visible; prev reads of other buf done
    if (i + 1 < NK) stage((i + 1) * 32, (i + 1) & 1);
    int b = i & 1;
    v16h a   = ldfragA(sA[b], fm, fk);
    v16h b10 = ldfragB(sB1[b], fn, fkb);
    v16h b11 = ldfragB(sB1[b], fn + 16, fkb);
    v16h b20 = ldfragB(sB2[b], fn, fkb);
    v16h b21 = ldfragB(sB2[b], fn + 16, fkb);
    aP0 = wmma16(a, b10, aP0);
    aP1 = wmma16(a, b11, aP1);
    aQ0 = wmma16(a, b20, aQ0);
    aQ1 = wmma16(a, b21, aQ1);
  }

  // epilogue: act = silu(h1) * gate
  int nlo = lane & 15;
  int mofs = (lane >> 4) * 8;
#pragma unroll
  for (int r = 0; r < 8; r++) {
    int row = r0 + mbase + mofs + r;
    if (row < cnt) {
      _Float16* dst = acts + (size_t)(base + row) * FF + f0 + nbase;
      dst[nlo]      = (_Float16)silu_gate(aP0[r], aQ0[r]);
      dst[16 + nlo] = (_Float16)silu_gate(aP1[r], aQ1[r]);
    }
  }
}

// ---------------------------------------------------------------------------
// K4: GEMM2  y[slot] = topw[slot] * (acts[slot] @ w2[e]^T). K-step 32 over F,
// double-buffered LDS.
// ---------------------------------------------------------------------------
__global__ __launch_bounds__(256) void k_gemm2(
    const _Float16* __restrict__ acts, const float* __restrict__ w2,
    const float* __restrict__ wslot, const int* __restrict__ segmeta,
    float* __restrict__ ybuf) {
  int e = blockIdx.y;
  int cnt = segmeta[EE + e];
  int r0 = blockIdx.z * 64;
  if (r0 >= cnt) return;
  int base = segmeta[e];
  int h0 = blockIdx.x * 64;

  __shared__ _Float16 sA[2][64][32];
  __shared__ _Float16 sB[2][64][32];

  int tid = threadIdx.x;
  int wv = tid >> 5, lane = tid & 31;
  int mbase = (wv & 3) * 16;
  int nbase = (wv >> 2) * 32;

  int arow = tid >> 2;
  int aseg = (tid & 3) * 8;
  int grow = r0 + arow;
  int crow = base + (grow < cnt ? grow : cnt - 1);
  const _Float16* asrc = acts + (size_t)crow * FF + aseg;
  unsigned aldst0 = (unsigned)(size_t)&sA[0][arow][aseg];
  unsigned aldst1 = (unsigned)(size_t)&sA[1][arow][aseg];

  int brow = tid >> 2;
  int bc = (tid & 3) * 8;
  const float* bsrc = w2 + ((size_t)e * HH + (h0 + brow)) * FF + bc;

  auto stage = [&](int k0, int buf) {
    async_copy16(buf ? aldst1 : aldst0, (uint64_t)(uintptr_t)(asrc + k0));
    const float4* p = (const float4*)(bsrc + k0);
    *(v8h*)&sB[buf][brow][bc] = cvt8(p[0], p[1]);
  };

  v8f a0 = {}, a1 = {};
  int fm = mbase + (lane & 15);
  int fk = (lane >> 4) * 8;
  int fn = nbase + (lane & 15);
  int fkb = (lane >> 4) * 8;

  const int NK = FF / 32;
  stage(0, 0);
  for (int i = 0; i < NK; ++i) {
    wait_async0();
    __syncthreads();
    if (i + 1 < NK) stage((i + 1) * 32, (i + 1) & 1);
    int b = i & 1;
    v16h a  = ldfragA(sA[b], fm, fk);
    v16h b0 = ldfragB(sB[b], fn, fkb);
    v16h b1 = ldfragB(sB[b], fn + 16, fkb);
    a0 = wmma16(a, b0, a0);
    a1 = wmma16(a, b1, a1);
  }

  int nlo = lane & 15;
  int mofs = (lane >> 4) * 8;
#pragma unroll
  for (int r = 0; r < 8; r++) {
    int row = r0 + mbase + mofs + r;
    if (row < cnt) {
      float s = wslot[base + row];
      float* dst = ybuf + (size_t)(base + row) * HH + h0 + nbase;
      dst[nlo] = a0[r] * s;
      dst[16 + nlo] = a1[r] * s;
    }
  }
}

// ---------------------------------------------------------------------------
// K5: combine the two scaled expert contributions per token (no atomics).
// ---------------------------------------------------------------------------
__global__ void k_combine(const float* __restrict__ ybuf,
                          const int* __restrict__ slotof,
                          float* __restrict__ out) {
  int i = (blockIdx.x * 256 + threadIdx.x) * 4;
  int t = i >> 11;              // / HH
  int h = i & (HH - 1);
  int s0 = slotof[t * 2], s1 = slotof[t * 2 + 1];
  float4 a = *(const float4*)(ybuf + (size_t)s0 * HH + h);
  float4 b = *(const float4*)(ybuf + (size_t)s1 * HH + h);
  float4 o = {a.x + b.x, a.y + b.y, a.z + b.z, a.w + b.w};
  *(float4*)(out + i) = o;
}

// ---------------------------------------------------------------------------
extern "C" void kernel_launch(void* const* d_in, const int* in_sizes, int n_in,
                              void* d_out, int out_size, void* d_ws, size_t ws_size,
                              hipStream_t stream) {
  (void)in_sizes; (void)n_in; (void)out_size; (void)ws_size;
  const float* x  = (const float*)d_in[0];
  const float* Wr = (const float*)d_in[1];
  const float* w1 = (const float*)d_in[2];
  const float* v1 = (const float*)d_in[3];
  const float* w2 = (const float*)d_in[4];
  float* out  = (float*)d_out;
  float* wout = out + (size_t)TT * HH;   // softmax weights output [T, E]

  char* ws = (char*)d_ws;
  _Float16* xh    = (_Float16*)(ws + OFF_XH);
  int*      topi  = (int*)(ws + OFF_TOPI);
  float*    topw  = (float*)(ws + OFF_TOPW);
  int*      glist = (int*)(ws + OFF_GLIST);
  float*    wslot = (float*)(ws + OFF_WSLOT);
  int*      slotof= (int*)(ws + OFF_SLOTOF);
  int*      segm  = (int*)(ws + OFF_SEG);
  _Float16* acts  = (_Float16*)(ws + OFF_ACTS);
  float*    ybuf  = (float*)(ws + OFF_YBUF);

  k_f2h<<<(TT * HH) / 1024, 256, 0, stream>>>(x, xh);
  k_router<<<TT / 8, 256, 0, stream>>>(x, Wr, wout, topi, topw);
  k_build<<<1, 256, 0, stream>>>(topi, topw, glist, wslot, slotof, segm);
  k_gemm1<<<dim3(FF / 64, EE, TT / 64), 256, 0, stream>>>(xh, w1, v1, glist, segm, acts);
  k_gemm2<<<dim3(HH / 64, EE, TT / 64), 256, 0, stream>>>(acts, w2, wslot, segm, ybuf);
  k_combine<<<(TT * HH) / 1024, 256, 0, stream>>>(ybuf, slotof, out);
}
